// MomGatedDeltaNet_70909910057148
// MI455X (gfx1250) — matrix operand order repaired and checked by reading, not compiled, not run on hardware
//
#include <hip/hip_runtime.h>
#include <hip/hip_bf16.h>

// ---------------- CDNA5 vector types ----------------
typedef __attribute__((ext_vector_type(16))) _Float16 v16h;
typedef __attribute__((ext_vector_type(8)))  _Float16 v8h;
typedef __attribute__((ext_vector_type(8)))  float    v8f;

// ---------------- problem constants ----------------
#define S_TOK   1024
#define DMODEL  2048
#define NEXP    8
#define NTOPK   2
#define NHEAD   6
#define DKH     256
#define DVH     512
#define KDIM    1536      // NHEAD*DKH
#define VDIM    3072      // NHEAD*DVH
#define CAPLEN  1024
#define NROWS   8192      // NEXP*CAPLEN

// ---------------- GEMM tiling ----------------
#define BM 128
#define BN 128
#define BK 32
#define LSTR 40           // padded LDS row stride in halves (80B, 16B aligned)

// ---------------- TDM availability ----------------
#if defined(__AMDGCN__) && __has_builtin(__builtin_amdgcn_tensor_load_to_lds)
#define USE_TDM 1
#else
#define USE_TDM 0
#endif

#if USE_TDM
typedef unsigned int __attribute__((ext_vector_type(4))) u32x4_;
typedef int __attribute__((ext_vector_type(8)))          i32x8_;
typedef int __attribute__((ext_vector_type(4)))          i32x4_;

// Issue a TDM 2D tile load: BM x BK f16 tile from row-major global (row
// stride K_ elements) into LDS at lds_addr, with LDS padding of 4 DWORDs
// every 16 DWORDs so rows land at the LSTR(=40 halves) stride the WMMA
// fragment readers use.  (D# layout per CDNA5 ISA)
__device__ __forceinline__ void tdm_load_tile(const _Float16* gptr, int K_,
                                              int M_, unsigned lds_addr) {
  unsigned long long ga = (unsigned long long)(size_t)gptr;
  u32x4_ g0;
  g0[0] = 1u;                                            // count=1, user desc
  g0[1] = lds_addr;                                      // lds_addr [63:32]
  g0[2] = (unsigned)(ga & 0xffffffffu);                  // global_addr lo
  g0[3] = (unsigned)((ga >> 32) & 0x01ffffffu) | (2u << 30); // ga hi | type=2
  unsigned td0 = (unsigned)K_;                           // tensor_dim0 (elems)
  unsigned td1 = (unsigned)M_;                           // tensor_dim1 (rows)
  unsigned long long st0 = (unsigned long long)(unsigned)K_; // dim0 stride
  i32x8_ g1;
  g1[0] = (int)((1u << 16)    // data_size = 1 -> 2 bytes
              | (1u << 20)    // pad_enable
              | (3u << 22)    // pad_interval: 16 DWORDs (64B) per row
              | (3u << 25));  // pad_amount: 4 DWORDs (16B) -> LSTR=40 halves
  g1[1] = (int)((td0 & 0xffffu) << 16);                  // td0[15:0] @ [63:48]
  g1[2] = (int)((td0 >> 16) | ((td1 & 0xffffu) << 16));  // td0 hi | td1 lo
  g1[3] = (int)((td1 >> 16) | ((unsigned)BK << 16));     // td1 hi | tile_dim0
  g1[4] = (int)(unsigned)BM;                             // tile_dim1, dim2=0
  g1[5] = (int)(unsigned)(st0 & 0xffffffffu);            // stride0 lo
  g1[6] = (int)(unsigned)((st0 >> 32) & 0xffffu);        // stride0 hi
  g1[7] = 0;
  i32x4_ gz = {0, 0, 0, 0};
#if __clang_major__ >= 23
  i32x8_ gz8 = {0, 0, 0, 0, 0, 0, 0, 0};
  __builtin_amdgcn_tensor_load_to_lds(g0, g1, gz, gz, gz8, 0);
#else
  __builtin_amdgcn_tensor_load_to_lds(g0, g1, gz, gz, 0);
#endif
}
#endif  // USE_TDM

// =====================================================================
// f32 -> f16 convert (weights / activations)
// =====================================================================
__global__ void cvt_f32_f16(const float* __restrict__ src,
                            _Float16* __restrict__ dst, long long n) {
  long long i = (long long)blockIdx.x * blockDim.x + threadIdx.x;
  long long stride = (long long)gridDim.x * blockDim.x;
  for (; i < n; i += stride) dst[i] = (_Float16)src[i];
}

// =====================================================================
// Router: softmax over 8 experts, top-2, renormalize
// =====================================================================
__global__ void router_kernel(const float* __restrict__ hs,
                              const float* __restrict__ gate_w,
                              float* __restrict__ rw, int* __restrict__ sel) {
  int s = blockIdx.x * blockDim.x + threadIdx.x;
  if (s >= S_TOK) return;
  float logit[NEXP];
  #pragma unroll
  for (int e = 0; e < NEXP; e++) logit[e] = 0.f;
  const float* x = hs + (long long)s * DMODEL;
  for (int d = 0; d < DMODEL; d++) {
    float xv = x[d];
    #pragma unroll
    for (int e = 0; e < NEXP; e++) logit[e] += xv * gate_w[d * NEXP + e];
  }
  float mx = logit[0];
  #pragma unroll
  for (int e = 1; e < NEXP; e++) mx = fmaxf(mx, logit[e]);
  float p[NEXP], sum = 0.f;
  #pragma unroll
  for (int e = 0; e < NEXP; e++) { p[e] = __expf(logit[e] - mx); sum += p[e]; }
  #pragma unroll
  for (int e = 0; e < NEXP; e++) p[e] /= sum;
  int i0 = 0;
  for (int e = 1; e < NEXP; e++) if (p[e] > p[i0]) i0 = e;
  int i1 = (i0 == 0) ? 1 : 0;
  for (int e = 0; e < NEXP; e++) if (e != i0 && p[e] > p[i1]) i1 = e;
  float ws = p[i0] + p[i1];
  rw[s * 2 + 0] = p[i0] / ws;
  rw[s * 2 + 1] = p[i1] / ws;
  sel[s * 2 + 0] = i0;
  sel[s * 2 + 1] = i1;
}

// =====================================================================
// Slot assignment: stable (temporal) order, right-aligned capacity buffer
// =====================================================================
__global__ void slot_kernel(const int* __restrict__ sel,
                            int* __restrict__ slot_token,
                            int* __restrict__ tok_slot,
                            int* __restrict__ counts) {
  int e = threadIdx.x;
  if (e >= NEXP) return;
  for (int i = 0; i < CAPLEN; i++) slot_token[e * CAPLEN + i] = -1;
  int c = 0;
  for (int s = 0; s < S_TOK; s++)
    for (int j = 0; j < NTOPK; j++)
      if (sel[s * 2 + j] == e) c++;
  counts[e] = c;
  int drop = (c > CAPLEN) ? (c - CAPLEN) : 0;
  int ck = c - drop;
  int base = CAPLEN - ck;      // right-aligned
  int pos = 0;
  for (int s = 0; s < S_TOK; s++)
    for (int j = 0; j < NTOPK; j++) {
      if (sel[s * 2 + j] != e) continue;
      if (pos >= drop) {
        int slot = base + (pos - drop);
        slot_token[e * CAPLEN + slot] = s;
        tok_slot[s * 2 + j] = e * CAPLEN + slot;
      } else {
        tok_slot[s * 2 + j] = -1;
      }
      pos++;
    }
}

// =====================================================================
// Dispatch: gather hidden rows into per-expert capacity buffers (f16)
// =====================================================================
__global__ void __launch_bounds__(256)
gather_kernel(const float* __restrict__ hs, const int* __restrict__ slot_token,
              _Float16* __restrict__ xe) {
  long long row = blockIdx.x;
  int tok = slot_token[row];
  int tid = threadIdx.x;
  _Float16* dst = xe + row * DMODEL;
  if (tok < 0) {
    for (int d = tid; d < DMODEL; d += 256) dst[d] = (_Float16)0.f;
  } else {
    const float* src = hs + (long long)tok * DMODEL;
    for (int d = tid; d < DMODEL; d += 256) dst[d] = (_Float16)src[d];
  }
}

// =====================================================================
// WMMA GEMM: C(f32, MxN) = A(f16, MxK, rm) * B(f16, KxN, rm)
// 128x128 block tile, BK=32, double-buffered LDS (1 barrier / K-step).
// A tile staged by the Tensor Data Mover (async DMA + LDS pad) when
// available; B staged through registers with transpose into LDS.
// Requires M%128==0, N%128==0, K%32==0 (true for all uses here).
// =====================================================================
__global__ void __launch_bounds__(256)
gemm_f16_wmma(const _Float16* __restrict__ A, const _Float16* __restrict__ B,
              float* __restrict__ C, int M, int N, int K,
              long long sA, long long sB, long long sC) {
  __shared__ _Float16 As[2][BM * LSTR];
  __shared__ _Float16 Bs[2][BN * LSTR];   // stored transposed: Bs[n][k]
  const int bz = blockIdx.z;
  A += (long long)bz * sA;
  B += (long long)bz * sB;
  C += (long long)bz * sC;
  const int m0 = blockIdx.y * BM;
  const int n0 = blockIdx.x * BN;
  const int tid  = threadIdx.x;
  const int lane = tid & 31;
  const int wave = tid >> 5;
  const int wm = wave >> 2;          // 0..1 -> 64 rows
  const int wn = wave & 3;           // 0..3 -> 32 cols
  const int hlf = lane >> 4;         // 0/1
  const int l16 = lane & 15;

  // ---- hoisted staging addresses ----
  const int rb0 = tid >> 4, cb = (tid & 15) * 8;         // B: 2 rows/thread
  const _Float16* bp0 = B + (long long)rb0 * N + n0 + cb;
  const _Float16* bp1 = bp0 + (long long)16 * N;
  const long long bstep = (long long)BK * N;
#if !USE_TDM
  const int ra0 = tid >> 2, ca = (tid & 3) * 8;          // A: 2 rows/thread
  const _Float16* ap0 = A + (long long)(m0 + ra0) * K + ca;
  const _Float16* ap1 = ap0 + (long long)64 * K;
#endif

  v8f acc[4][2] = {};
  v8h rB0, rB1;
#if !USE_TDM
  v8h rA0, rA1;
#endif

  const int nsteps = K / BK;

  // ---- prologue: stage tile 0 into buffer 0 ----
#if USE_TDM
  if (tid < 32)
    tdm_load_tile(A + (long long)m0 * K, K, M, (unsigned)(size_t)&As[0][0]);
#else
  rA0 = *(const v8h*)(ap0);
  rA1 = *(const v8h*)(ap1);
  *(v8h*)(&As[0][ra0 * LSTR + ca]) = rA0;
  *(v8h*)(&As[0][(ra0 + 64) * LSTR + ca]) = rA1;
#endif
  rB0 = *(const v8h*)(bp0);
  rB1 = *(const v8h*)(bp1);
  #pragma unroll
  for (int j = 0; j < 8; j++) {
    Bs[0][(cb + j) * LSTR + rb0] = rB0[j];
    Bs[0][(cb + j) * LSTR + rb0 + 16] = rB1[j];
  }

  int cur = 0;
  for (int step = 0; step < nsteps; ++step) {
    const bool has_next = (step + 1) < nsteps;
    const int ko_next = (step + 1) * BK;
#if USE_TDM
    if (tid < 32) __builtin_amdgcn_s_wait_tensorcnt(0);  // A[cur] DMA done
#endif
    __syncthreads();                                     // A[cur],B[cur] ready

    // ---- issue fetches for next tile (overlap with WMMA below) ----
    if (has_next) {
#if USE_TDM
      if (tid < 32)
        tdm_load_tile(A + (long long)m0 * K + ko_next, K, M,
                      (unsigned)(size_t)&As[cur ^ 1][0]);
#else
      rA0 = *(const v8h*)(ap0 + ko_next);
      rA1 = *(const v8h*)(ap1 + ko_next);
#endif
      rB0 = *(const v8h*)(bp0 + (long long)(step + 1) * bstep);
      rB1 = *(const v8h*)(bp1 + (long long)(step + 1) * bstep);
    }

    // ---- fragments (ISA 7.12.2 layouts) + 8 WMMAs ----
    const _Float16* Ab = &As[cur][0];
    const _Float16* Bb = &Bs[cur][0];
    v16h afrag[4], bfrag[2];
    #pragma unroll
    for (int mi = 0; mi < 4; mi++) {
      int r = wm * 64 + mi * 16 + l16;
      v8h lo = *(const v8h*)(&Ab[r * LSTR + hlf * 8]);
      v8h hi = *(const v8h*)(&Ab[r * LSTR + 16 + hlf * 8]);
      #pragma unroll
      for (int j = 0; j < 8; j++) { afrag[mi][j] = lo[j]; afrag[mi][8 + j] = hi[j]; }
    }
    #pragma unroll
    for (int ni = 0; ni < 2; ni++) {
      int cc = wn * 32 + ni * 16 + l16;
      v8h lo = *(const v8h*)(&Bb[cc * LSTR + hlf * 8]);
      v8h hi = *(const v8h*)(&Bb[cc * LSTR + 16 + hlf * 8]);
      #pragma unroll
      for (int j = 0; j < 8; j++) { bfrag[ni][j] = lo[j]; bfrag[ni][8 + j] = hi[j]; }
    }
    #pragma unroll
    for (int mi = 0; mi < 4; mi++)
      #pragma unroll
      for (int ni = 0; ni < 2; ni++)
        acc[mi][ni] = __builtin_amdgcn_wmma_f32_16x16x32_f16(
            false, afrag[mi], false, bfrag[ni], (short)0, acc[mi][ni],
            false, false);

    // ---- drain staged regs into the other buffer (no barrier needed:
    //      nobody reads buf^1 until the barrier at the top of next step) ----
    if (has_next) {
#if !USE_TDM
      *(v8h*)(&As[cur ^ 1][ra0 * LSTR + ca]) = rA0;
      *(v8h*)(&As[cur ^ 1][(ra0 + 64) * LSTR + ca]) = rA1;
#endif
      #pragma unroll
      for (int j = 0; j < 8; j++) {
        Bs[cur ^ 1][(cb + j) * LSTR + rb0] = rB0[j];
        Bs[cur ^ 1][(cb + j) * LSTR + rb0 + 16] = rB1[j];
      }
    }
    cur ^= 1;
  }

  // ---- epilogue: C/D layout VGPR j -> M = 8*hlf + j ----
  #pragma unroll
  for (int mi = 0; mi < 4; mi++)
    #pragma unroll
    for (int ni = 0; ni < 2; ni++) {
      int col = n0 + wn * 32 + ni * 16 + l16;
      #pragma unroll
      for (int j = 0; j < 8; j++) {
        int row = m0 + wm * 64 + mi * 16 + hlf * 8 + j;
        C[(long long)row * N + col] = acc[mi][ni][j];
      }
    }
}

// =====================================================================
// beta / decay head projections (H=6, tiny) + activations
// =====================================================================
__global__ void __launch_bounds__(256)
betaa_kernel(const float* __restrict__ hs, const int* __restrict__ slot_token,
             const float* __restrict__ b_w, const float* __restrict__ a_w,
             const float* __restrict__ A_log, const float* __restrict__ dt_bias,
             float* __restrict__ beta, float* __restrict__ gdec) {
  __shared__ float red[12][256];
  long long row = blockIdx.x;          // e*CAPLEN + slot
  int e = (int)(row >> 10);
  int tok = slot_token[row];
  int tid = threadIdx.x;
  float accb[NHEAD] = {}, acca[NHEAD] = {};
  if (tok >= 0) {
    const float* x  = hs + (long long)tok * DMODEL;
    const float* bw = b_w + (long long)e * DMODEL * NHEAD;
    const float* aw = a_w + (long long)e * DMODEL * NHEAD;
    for (int d = tid; d < DMODEL; d += 256) {
      float xv = x[d];
      #pragma unroll
      for (int h = 0; h < NHEAD; h++) {
        accb[h] += xv * bw[d * NHEAD + h];
        acca[h] += xv * aw[d * NHEAD + h];
      }
    }
  }
  #pragma unroll
  for (int h = 0; h < NHEAD; h++) { red[h][tid] = accb[h]; red[6 + h][tid] = acca[h]; }
  __syncthreads();
  for (int st = 128; st > 0; st >>= 1) {
    if (tid < st)
      #pragma unroll
      for (int j = 0; j < 12; j++) red[j][tid] += red[j][tid + st];
    __syncthreads();
  }
  if (tid < NHEAD) {
    float bsum = red[tid][0], asum = red[6 + tid][0];
    float bt = 1.f / (1.f + __expf(-bsum));
    float z = asum + dt_bias[tid];
    float sp = (z > 20.f) ? z : log1pf(__expf(z));
    beta[row * NHEAD + tid] = bt;
    gdec[row * NHEAD + tid] = -__expf(A_log[tid]) * sp;
  }
}

// =====================================================================
// Depthwise causal conv(K=4) + silu + per-head l2-norm (q and k)
// =====================================================================
__global__ void __launch_bounds__(256)
conv_norm_kernel(const float* __restrict__ pre, const float* __restrict__ cw,
                 float* __restrict__ post, float scale) {
  __shared__ float red[256];
  long long nt = blockIdx.x;
  int h = blockIdx.y;
  int t = (int)(nt & (CAPLEN - 1));
  int tid = threadIdx.x;
  int c = h * DKH + tid;
  long long nbase = nt - t;
  float y = 0.f;
  #pragma unroll
  for (int j = 0; j < 4; j++) {
    int tt = t - 3 + j;
    if (tt >= 0) y += pre[(nbase + tt) * KDIM + c] * cw[c * 4 + j];
  }
  y = y / (1.f + __expf(-y));          // silu
  red[tid] = y * y;
  __syncthreads();
  for (int st = 128; st > 0; st >>= 1) {
    if (tid < st) red[tid] += red[tid + st];
    __syncthreads();
  }
  float rs = rsqrtf(red[0] + 1e-6f);
  post[nt * KDIM + c] = y * rs * scale;
}

// =====================================================================
// Depthwise causal conv(K=4) + silu for v (no norm)
// =====================================================================
__global__ void __launch_bounds__(256)
conv_v_kernel(const float* __restrict__ pre, const float* __restrict__ cw,
              float* __restrict__ post) {
  long long nt = blockIdx.x;
  int t = (int)(nt & (CAPLEN - 1));
  int tid = threadIdx.x;
  long long nbase = nt - t;
  for (int c = tid; c < VDIM; c += 256) {
    float y = 0.f;
    #pragma unroll
    for (int j = 0; j < 4; j++) {
      int tt = t - 3 + j;
      if (tt >= 0) y += pre[(nbase + tt) * VDIM + c] * cw[c * 4 + j];
    }
    post[nt * VDIM + c] = y / (1.f + __expf(-y));
  }
}

// =====================================================================
// Gated delta-rule recurrence: one block per (n, h, dv-half); the
// 256x256 f32 state slice is LDS-resident (256KB of CDNA5's 320KB WGP LDS).
// Decay exp(g) fused into the two state passes.
// =====================================================================
__global__ void __launch_bounds__(256)
recur_kernel(const float* __restrict__ qp, const float* __restrict__ kp,
             const float* __restrict__ vp, const float* __restrict__ beta,
             const float* __restrict__ gdec, float* __restrict__ obuf) {
  extern __shared__ float smem[];
  float* S  = smem;                    // 256 * 256 state
  float* kb = smem + 256 * 256;        // 256
  float* qb = kb + 256;                // 256
  const int tid  = threadIdx.x;        // owns one v column
  const int hv   = blockIdx.x & 1;     // dv half
  const int h    = (blockIdx.x >> 1) % NHEAD;
  const int n    = blockIdx.x / (2 * NHEAD);
  for (int k = 0; k < 256; k++) S[k * 256 + tid] = 0.f;
  for (int t = 0; t < CAPLEN; t++) {
    long long nt = (long long)n * CAPLEN + t;
    long long rq = nt * KDIM + (long long)h * DKH;
    long long rv = nt * VDIM + (long long)h * DVH + hv * 256;
    kb[tid] = kp[rq + tid];
    qb[tid] = qp[rq + tid];
    float vt = vp[rv + tid];
    float bt = beta[nt * NHEAD + h];
    float gt = gdec[nt * NHEAD + h];
    __syncthreads();
    float eg = __expf(gt);
    float pred = 0.f;
    #pragma unroll 8
    for (int k = 0; k < 256; k++) pred += kb[k] * S[k * 256 + tid];
    pred *= eg;
    float delta = (vt - pred) * bt;
    float o = 0.f;
    #pragma unroll 8
    for (int k = 0; k < 256; k++) {
      float sv = S[k * 256 + tid] * eg + kb[k] * delta;
      S[k * 256 + tid] = sv;
      o += qb[k] * sv;
    }
    obuf[rv + tid] = o;
    __syncthreads();
  }
}

// =====================================================================
// Combine (scatter-back with routing weights) + gated RMSNorm -> f16
// =====================================================================
__global__ void __launch_bounds__(256)
combine_kernel(const float* __restrict__ obuf, const float* __restrict__ rw,
               const int* __restrict__ tok_slot, const float* __restrict__ gg,
               const float* __restrict__ norm_w, _Float16* __restrict__ out_pre) {
  __shared__ float red[256];
  __shared__ float ovals[DVH];
  int s = blockIdx.x;
  int h = blockIdx.y;
  int tid = threadIdx.x;
  int sl0 = tok_slot[s * 2 + 0], sl1 = tok_slot[s * 2 + 1];
  float w0 = rw[s * 2 + 0], w1 = rw[s * 2 + 1];
  float ss = 0.f;
  for (int dv = tid; dv < DVH; dv += 256) {
    float o = 0.f;
    if (sl0 >= 0) o += w0 * obuf[((long long)sl0 * NHEAD + h) * DVH + dv];
    if (sl1 >= 0) o += w1 * obuf[((long long)sl1 * NHEAD + h) * DVH + dv];
    ovals[dv] = o;
    ss += o * o;
  }
  red[tid] = ss;
  __syncthreads();
  for (int st = 128; st > 0; st >>= 1) {
    if (tid < st) red[tid] += red[tid + st];
    __syncthreads();
  }
  float rms = rsqrtf(red[0] / (float)DVH + 1e-5f);
  for (int dv = tid; dv < DVH; dv += 256) {
    float gv = gg[(long long)s * VDIM + h * DVH + dv];
    float val = ovals[dv] * rms * norm_w[dv] * (gv / (1.f + __expf(-gv)));
    out_pre[(long long)s * VDIM + h * DVH + dv] = (_Float16)val;
  }
}

// =====================================================================
// Host launcher
// =====================================================================
extern "C" void kernel_launch(void* const* d_in, const int* in_sizes, int n_in,
                              void* d_out, int out_size, void* d_ws, size_t ws_size,
                              hipStream_t stream) {
  (void)in_sizes; (void)n_in; (void)out_size; (void)ws_size;
  const float* hs       = (const float*)d_in[0];
  const float* q_w      = (const float*)d_in[1];
  const float* gate_w   = (const float*)d_in[2];
  const float* k_w      = (const float*)d_in[3];
  const float* v_w      = (const float*)d_in[4];
  const float* b_w      = (const float*)d_in[5];
  const float* a_w      = (const float*)d_in[6];
  const float* A_log    = (const float*)d_in[7];
  const float* dt_bias  = (const float*)d_in[8];
  const float* q_conv_w = (const float*)d_in[9];
  const float* k_conv_w = (const float*)d_in[10];
  const float* v_conv_w = (const float*)d_in[11];
  const float* g_w      = (const float*)d_in[12];
  const float* norm_w   = (const float*)d_in[13];
  const float* o_w      = (const float*)d_in[14];
  float* out = (float*)d_out;

  size_t off = 0;
  auto alloc = [&](size_t bytes) {
    void* p = (char*)d_ws + off;
    off += (bytes + 255) & ~(size_t)255;
    return p;
  };
  _Float16* hidden16 = (_Float16*)alloc((size_t)S_TOK * DMODEL * 2);
  _Float16* qw16     = (_Float16*)alloc((size_t)DMODEL * KDIM * 2);
  _Float16* kw16     = (_Float16*)alloc((size_t)NEXP * DMODEL * KDIM * 2);
  _Float16* vw16     = (_Float16*)alloc((size_t)NEXP * DMODEL * VDIM * 2);
  _Float16* gw16     = (_Float16*)alloc((size_t)DMODEL * VDIM * 2);
  _Float16* ow16     = (_Float16*)alloc((size_t)VDIM * DMODEL * 2);
  _Float16* xe16     = (_Float16*)alloc((size_t)NROWS * DMODEL * 2);
  float* q_pre  = (float*)alloc((size_t)NROWS * KDIM * 4);
  float* k_pre  = (float*)alloc((size_t)NROWS * KDIM * 4);
  float* v_pre  = (float*)alloc((size_t)NROWS * VDIM * 4);
  float* q_post = (float*)alloc((size_t)NROWS * KDIM * 4);
  float* k_post = (float*)alloc((size_t)NROWS * KDIM * 4);
  float* v_post = (float*)alloc((size_t)NROWS * VDIM * 4);
  float* gg     = (float*)alloc((size_t)S_TOK * VDIM * 4);
  float* obuf   = (float*)alloc((size_t)NROWS * VDIM * 4);
  _Float16* outpre16 = (_Float16*)alloc((size_t)S_TOK * VDIM * 2);
  float* beta  = (float*)alloc((size_t)NROWS * NHEAD * 4);
  float* gdec  = (float*)alloc((size_t)NROWS * NHEAD * 4);
  float* rwb   = (float*)alloc((size_t)S_TOK * 2 * 4);
  int* sel        = (int*)alloc((size_t)S_TOK * 2 * 4);
  int* slot_token = (int*)alloc((size_t)NROWS * 4);
  int* tok_slot   = (int*)alloc((size_t)S_TOK * 2 * 4);
  int* counts     = (int*)alloc((size_t)NEXP * 4);

  // 1) precision staging (f32 -> f16)
  cvt_f32_f16<<<1024, 256, 0, stream>>>(hs, hidden16, (long long)S_TOK * DMODEL);
  cvt_f32_f16<<<1024, 256, 0, stream>>>(q_w, qw16, (long long)DMODEL * KDIM);
  cvt_f32_f16<<<2048, 256, 0, stream>>>(k_w, kw16, (long long)NEXP * DMODEL * KDIM);
  cvt_f32_f16<<<2048, 256, 0, stream>>>(v_w, vw16, (long long)NEXP * DMODEL * VDIM);
  cvt_f32_f16<<<1024, 256, 0, stream>>>(g_w, gw16, (long long)DMODEL * VDIM);
  cvt_f32_f16<<<1024, 256, 0, stream>>>(o_w, ow16, (long long)VDIM * DMODEL);

  // 2) router + slotting + dispatch
  router_kernel<<<4, 256, 0, stream>>>(hs, gate_w, rwb, sel);
  slot_kernel<<<1, 32, 0, stream>>>(sel, slot_token, tok_slot, counts);
  gather_kernel<<<NROWS, 256, 0, stream>>>(hs, slot_token, xe16);

  // 3) WMMA GEMMs: q (shared weight, M=8192), k/v (batched per expert)
  gemm_f16_wmma<<<dim3(KDIM / BN, NROWS / BM, 1), 256, 0, stream>>>(
      xe16, qw16, q_pre, NROWS, KDIM, DMODEL, 0, 0, 0);
  gemm_f16_wmma<<<dim3(KDIM / BN, CAPLEN / BM, NEXP), 256, 0, stream>>>(
      xe16, kw16, k_pre, CAPLEN, KDIM, DMODEL,
      (long long)CAPLEN * DMODEL, (long long)DMODEL * KDIM, (long long)CAPLEN * KDIM);
  gemm_f16_wmma<<<dim3(VDIM / BN, CAPLEN / BM, NEXP), 256, 0, stream>>>(
      xe16, vw16, v_pre, CAPLEN, VDIM, DMODEL,
      (long long)CAPLEN * DMODEL, (long long)DMODEL * VDIM, (long long)CAPLEN * VDIM);

  // 4) beta / decay projections
  betaa_kernel<<<NROWS, 256, 0, stream>>>(hs, slot_token, b_w, a_w, A_log,
                                          dt_bias, beta, gdec);

  // 5) conv + silu (+ l2norm for q/k; q also scaled by DK^-1/2)
  conv_norm_kernel<<<dim3(NROWS, NHEAD), 256, 0, stream>>>(q_pre, q_conv_w, q_post, 0.0625f);
  conv_norm_kernel<<<dim3(NROWS, NHEAD), 256, 0, stream>>>(k_pre, k_conv_w, k_post, 1.0f);
  conv_v_kernel<<<NROWS, 256, 0, stream>>>(v_pre, v_conv_w, v_post);

  // 6) gated delta-rule scan (LDS-resident state)
  const int smem_bytes = 256 * 256 * 4 + 2 * 256 * 4;
  hipFuncSetAttribute(reinterpret_cast<const void*>(recur_kernel),
                      hipFuncAttributeMaxDynamicSharedMemorySize, smem_bytes);
  recur_kernel<<<NEXP * NHEAD * 2, 256, smem_bytes, stream>>>(
      q_post, k_post, v_post, beta, gdec, obuf);

  // 7) gate projection gg = hidden @ g_w (WMMA), combine + RMSNorm
  gemm_f16_wmma<<<dim3(VDIM / BN, S_TOK / BM, 1), 256, 0, stream>>>(
      hidden16, gw16, gg, S_TOK, VDIM, DMODEL, 0, 0, 0);
  combine_kernel<<<dim3(S_TOK, NHEAD), 256, 0, stream>>>(
      obuf, rwb, tok_slot, gg, norm_w, outpre16);

  // 8) output projection (WMMA) -> d_out (f32)
  gemm_f16_wmma<<<dim3(DMODEL / BN, S_TOK / BM, 1), 256, 0, stream>>>(
      outpre16, ow16, out, S_TOK, DMODEL, VDIM, 0, 0, 0);
}
